// PointCloudProjector_16569983828653
// MI455X (gfx1250) — matrix operand order repaired and checked by reading, not compiled, not run on hardware
//
#include <hip/hip_runtime.h>
#include <hip/hip_bf16.h>
#include <math.h>

#define NPTS   262144
#define BATCH  32
#define KSEL   256
#define HID    768
#define INDIM  60
#define KPAD   64
#define NFREQ  10
#define CHUNKS 8
#define CHUNK_PTS (NPTS / CHUNKS)   // 32768 points per block

typedef __attribute__((ext_vector_type(16))) _Float16 v16h;
typedef __attribute__((ext_vector_type(8)))  _Float16 v8h;
typedef __attribute__((ext_vector_type(8)))  float    v8f;

__device__ __forceinline__ float sanitize(float v) {
    return __builtin_isfinite(v) ? v : 0.0f;
}

// Block-wide argmax (1024 threads) with first-index tie-break.
__device__ __forceinline__ void block_argmax(float* s_val, int* s_idx, int tid) {
    for (int s = 512; s > 0; s >>= 1) {
        if (tid < s) {
            float ov = s_val[tid + s]; int oi = s_idx[tid + s];
            float mv = s_val[tid];     int mi = s_idx[tid];
            if (ov > mv || (ov == mv && oi < mi)) { s_val[tid] = ov; s_idx[tid] = oi; }
        }
        __syncthreads();
    }
}

// ------------------------------------------------------------------
// FPS iteration 1: dist = |p - p0|^2 over this block's chunk; emit
// per-block argmax candidate. 8 blocks per batch -> 256 blocks active.
// ------------------------------------------------------------------
__global__ __launch_bounds__(1024)
void fps_init(const float* __restrict__ pc, float* __restrict__ dists,
              float* __restrict__ sampled, float* __restrict__ candval,
              int* __restrict__ candidx) {
    const int chunk = blockIdx.x;              // 0..7
    const int b     = blockIdx.y;              // 0..31
    const int tid   = threadIdx.x;
    const float* P  = pc + (size_t)b * NPTS * 3;
    float* dist     = dists + (size_t)b * NPTS;

    __shared__ float s_val[1024];
    __shared__ int   s_idx[1024];

    const float lx = sanitize(P[0]);
    const float ly = sanitize(P[1]);
    const float lz = sanitize(P[2]);
    if (chunk == 0 && tid == 0) {
        float* out = sampled + (size_t)b * KSEL * 3;
        out[0] = lx; out[1] = ly; out[2] = lz;
    }

    const int base = chunk * CHUNK_PTS;
    float best = -1.0f; int bidx = base;
    #pragma unroll 4
    for (int j = base + tid; j < base + CHUNK_PTS; j += 1024) {
        const float* pj = P + 3 * j;
        float dx = sanitize(pj[0]) - lx;
        float dy = sanitize(pj[1]) - ly;
        float dz = sanitize(pj[2]) - lz;
        float d = dx*dx + dy*dy + dz*dz;
        dist[j] = d;
        if (d > best) { best = d; bidx = j; }
    }
    s_val[tid] = best; s_idx[tid] = bidx;
    __syncthreads();
    block_argmax(s_val, s_idx, tid);
    if (tid == 0) {
        candval[b * CHUNKS + chunk] = s_val[0];
        candidx[b * CHUNKS + chunk] = s_idx[0];
    }
}

// ------------------------------------------------------------------
// FPS iteration i (2..255): reduce previous candidates (8 tiny reads),
// write sampled[i-1], min-update dists over chunk, emit new candidate.
// Cross-launch ordering on the stream provides the global barrier.
// ------------------------------------------------------------------
__global__ __launch_bounds__(1024)
void fps_step(const float* __restrict__ pc, float* __restrict__ dists,
              float* __restrict__ sampled, float* __restrict__ candval,
              int* __restrict__ candidx, int i) {
    const int chunk = blockIdx.x;
    const int b     = blockIdx.y;
    const int tid   = threadIdx.x;
    const float* P  = pc + (size_t)b * NPTS * 3;
    float* dist     = dists + (size_t)b * NPTS;

    __shared__ float s_val[1024];
    __shared__ int   s_idx[1024];
    __shared__ float s_last[3];

    if (tid == 0) {
        float bv = candval[b * CHUNKS]; int bi = candidx[b * CHUNKS];
        #pragma unroll
        for (int c = 1; c < CHUNKS; ++c) {
            float v = candval[b * CHUNKS + c];
            int  ix = candidx[b * CHUNKS + c];
            if (v > bv || (v == bv && ix < bi)) { bv = v; bi = ix; }
        }
        const float* pj = P + 3 * bi;
        float x = sanitize(pj[0]), y = sanitize(pj[1]), z = sanitize(pj[2]);
        s_last[0] = x; s_last[1] = y; s_last[2] = z;
        if (chunk == 0) {
            float* out = sampled + (size_t)b * KSEL * 3 + (size_t)(i - 1) * 3;
            out[0] = x; out[1] = y; out[2] = z;
        }
    }
    __syncthreads();

    const float lx = s_last[0], ly = s_last[1], lz = s_last[2];
    const int base = chunk * CHUNK_PTS;
    float best = -1.0f; int bidx = base;
    #pragma unroll 4
    for (int j = base + tid; j < base + CHUNK_PTS; j += 1024) {
        const float* pj = P + 3 * j;
        float dx = sanitize(pj[0]) - lx;
        float dy = sanitize(pj[1]) - ly;
        float dz = sanitize(pj[2]) - lz;
        float d = dx*dx + dy*dy + dz*dz;
        float dm = dist[j];
        dm = (d < dm) ? d : dm;
        dist[j] = dm;
        if (dm > best) { best = dm; bidx = j; }
    }
    s_val[tid] = best; s_idx[tid] = bidx;
    __syncthreads();
    block_argmax(s_val, s_idx, tid);
    if (tid == 0) {
        candval[b * CHUNKS + chunk] = s_val[0];
        candidx[b * CHUNKS + chunk] = s_idx[0];
    }
}

// ------------------------------------------------------------------
// Resolve the final (i=255) candidate in-block, then normalize
// (centroid + bbox scale) and Fourier-encode to f16 rows [B*256][64]
// (K padded 60->64 with zeros), laid out for WMMA A-fragments.
// ------------------------------------------------------------------
__global__ __launch_bounds__(256)
void encode_kernel(const float* __restrict__ pc, const float* __restrict__ sampled,
                   const float* __restrict__ candval, const int* __restrict__ candidx,
                   _Float16* __restrict__ encA) {
    const int b   = blockIdx.x;
    const int tid = threadIdx.x;
    const float* S = sampled + (size_t)b * KSEL * 3;

    __shared__ float red[256];
    __shared__ float stat[3];
    __shared__ float s_p255[3];

    if (tid == 0) {   // finalize FPS: winner of iteration 255
        float bv = candval[b * CHUNKS]; int bi = candidx[b * CHUNKS];
        #pragma unroll
        for (int c = 1; c < CHUNKS; ++c) {
            float v = candval[b * CHUNKS + c];
            int  ix = candidx[b * CHUNKS + c];
            if (v > bv || (v == bv && ix < bi)) { bv = v; bi = ix; }
        }
        const float* pj = pc + (size_t)b * NPTS * 3 + 3 * bi;
        s_p255[0] = sanitize(pj[0]);
        s_p255[1] = sanitize(pj[1]);
        s_p255[2] = sanitize(pj[2]);
    }
    __syncthreads();

    float px, py, pz;
    if (tid == KSEL - 1) { px = s_p255[0]; py = s_p255[1]; pz = s_p255[2]; }
    else                 { px = S[3*tid+0]; py = S[3*tid+1]; pz = S[3*tid+2]; }

    float vals[3] = {px, py, pz};
    for (int c = 0; c < 3; ++c) {
        red[tid] = vals[c]; __syncthreads();
        for (int s = 128; s > 0; s >>= 1) {
            if (tid < s) red[tid] += red[tid + s];
            __syncthreads();
        }
        if (tid == 0) stat[c] = red[0] * (1.0f / 256.0f);
        __syncthreads();
    }
    float ox = px - stat[0], oy = py - stat[1], oz = pz - stat[2];

    float cen[3] = {ox, oy, oz};
    float maxdim = -INFINITY;
    for (int c = 0; c < 3; ++c) {
        red[tid] = cen[c]; __syncthreads();
        for (int s = 128; s > 0; s >>= 1) {
            if (tid < s) red[tid] = fmaxf(red[tid], red[tid + s]);
            __syncthreads();
        }
        float mx = red[0]; __syncthreads();
        red[tid] = cen[c]; __syncthreads();
        for (int s = 128; s > 0; s >>= 1) {
            if (tid < s) red[tid] = fminf(red[tid], red[tid + s]);
            __syncthreads();
        }
        float mn = red[0]; __syncthreads();
        maxdim = fmaxf(maxdim, mx - mn);
    }
    float scale = (maxdim > 1e-8f) ? maxdim : 1.0f;
    float inv = 1.0f / scale;

    const float PI = 3.14159265358979323846f;
    float nrm[3] = {ox * inv, oy * inv, oz * inv};
    _Float16* row = encA + (size_t)(b * KSEL + tid) * KPAD;
    for (int c = 0; c < 3; ++c) {
        float ang = nrm[c] * PI;            // freq 2^0
        #pragma unroll
        for (int k = 0; k < NFREQ; ++k) {
            float sv, cv;
            __sincosf(ang, &sv, &cv);
            row[c*2*NFREQ + 2*k + 0] = (_Float16)sv;
            row[c*2*NFREQ + 2*k + 1] = (_Float16)cv;
            ang *= 2.0f;
        }
    }
    row[60] = (_Float16)0.0f; row[61] = (_Float16)0.0f;
    row[62] = (_Float16)0.0f; row[63] = (_Float16)0.0f;
}

// ------------------------------------------------------------------
// Pad/convert W [768][60] f32 -> [768][64] f16 (rows = output channel,
// i.e. already B = W^T for D = A x B with A rows = points).
// ------------------------------------------------------------------
__global__ __launch_bounds__(256)
void wpad_kernel(const float* __restrict__ W, _Float16* __restrict__ Wp) {
    int idx = blockIdx.x * 256 + threadIdx.x;
    if (idx >= HID * KPAD) return;
    int n = idx / KPAD, k = idx % KPAD;
    float v = (k < INDIM) ? W[n * INDIM + k] : 0.0f;
    Wp[idx] = (_Float16)v;
}

// ------------------------------------------------------------------
// GEMM: out[M=8192][N=768] = enc[M][64] x Wp^T + bias, via
// v_wmma_f32_16x16x32_f16 (2 per tile, K=64). One wave per 16x16 tile.
// Fragment layouts per CDNA5 ISA 7.12.2.
// ------------------------------------------------------------------
__global__ __launch_bounds__(256)
void gemm_kernel(const _Float16* __restrict__ encA, const _Float16* __restrict__ Wp,
                 const float* __restrict__ bias, float* __restrict__ out) {
    const int lane  = threadIdx.x & 31;
    const int wave  = threadIdx.x >> 5;
    const int ntile = blockIdx.x;               // 0..47
    const int mtile = blockIdx.y * 8 + wave;    // 0..511
    const int g  = lane >> 4;                   // lane group 0/1
    const int ml = lane & 15;

    // A fragments: row M = mtile*16 + ml; K blocks per ISA A-layout
    const _Float16* arow = encA + (size_t)(mtile * 16 + ml) * KPAD;
    v16h a0, a1;
    {
        v8h lo0 = *(const v8h*)(arow + 8 * g);        // K =      8g .. 8g+7
        v8h hi0 = *(const v8h*)(arow + 16 + 8 * g);   // K = 16 + 8g ..
        v8h lo1 = *(const v8h*)(arow + 32 + 8 * g);   // K = 32 + 8g ..
        v8h hi1 = *(const v8h*)(arow + 48 + 8 * g);   // K = 48 + 8g ..
        #pragma unroll
        for (int e = 0; e < 8; ++e) {
            a0[e] = lo0[e]; a0[8 + e] = hi0[e];
            a1[e] = lo1[e]; a1[8 + e] = hi1[e];
        }
    }
    // B fragments: col N = ntile*16 + ml; K blocks per ISA layout (same as A)
    const _Float16* brow = Wp + (size_t)(ntile * 16 + ml) * KPAD;
    v16h b0 = __builtin_shufflevector(*(const v8h*)(brow + 8 * g), *(const v8h*)(brow + 16 + 8 * g), 0, 1, 2, 3, 4, 5, 6, 7, 8, 9, 10, 11, 12, 13, 14, 15);
    v16h b1 = __builtin_shufflevector(*(const v8h*)(brow + 32 + 8 * g), *(const v8h*)(brow + 48 + 8 * g), 0, 1, 2, 3, 4, 5, 6, 7, 8, 9, 10, 11, 12, 13, 14, 15);

    v8f c = {};
    c = __builtin_amdgcn_wmma_f32_16x16x32_f16(false, a0, false, b0, (short)0, c, false, false);
    c = __builtin_amdgcn_wmma_f32_16x16x32_f16(false, a1, false, b1, (short)0, c, false, false);

    const int ncol = ntile * 16 + ml;
    const float bb = bias[ncol];
    #pragma unroll
    for (int r = 0; r < 8; ++r) {
        int M = mtile * 16 + r + 8 * g;       // C/D layout: VGPR r, lanes 16-31 -> M+8
        out[(size_t)M * HID + ncol] = c[r] + bb;
    }
}

extern "C" void kernel_launch(void* const* d_in, const int* in_sizes, int n_in,
                              void* d_out, int out_size, void* d_ws, size_t ws_size,
                              hipStream_t stream) {
    const float* pc   = (const float*)d_in[0];   // [32][262144][3] f32
    const float* W    = (const float*)d_in[1];   // [768][60] f32
    const float* bias = (const float*)d_in[2];   // [768] f32
    float* out = (float*)d_out;                  // [32][256][768] f32

    char* ws = (char*)d_ws;
    size_t off = 0;
    float* dists = (float*)(ws + off);      off += (size_t)BATCH * NPTS * sizeof(float);      // 32 MB
    float* sampled = (float*)(ws + off);    off += (size_t)BATCH * KSEL * 3 * sizeof(float);  // 96 KB
    _Float16* encA = (_Float16*)(ws + off); off += (size_t)BATCH * KSEL * KPAD * 2;           // 1 MB
    _Float16* Wp   = (_Float16*)(ws + off); off += (size_t)HID * KPAD * 2;                    // 96 KB
    float* candval = (float*)(ws + off);    off += (size_t)BATCH * CHUNKS * sizeof(float);
    int*   candidx = (int*)(ws + off);      off += (size_t)BATCH * CHUNKS * sizeof(int);

    const dim3 fps_grid(CHUNKS, BATCH);
    fps_init<<<fps_grid, 1024, 0, stream>>>(pc, dists, sampled, candval, candidx);
    for (int i = 2; i < KSEL; ++i) {
        fps_step<<<fps_grid, 1024, 0, stream>>>(pc, dists, sampled, candval, candidx, i);
    }
    // i = KSEL winner is resolved inside encode_kernel
    encode_kernel<<<BATCH, 256, 0, stream>>>(pc, sampled, candval, candidx, encA);
    wpad_kernel<<<(HID * KPAD + 255) / 256, 256, 0, stream>>>(W, Wp);
    gemm_kernel<<<dim3(HID / 16, 64), 256, 0, stream>>>(encA, Wp, bias, out);
}
